// EquinoxGRUWrapper_65017214926966
// MI455X (gfx1250) — matrix operand order repaired and checked by reading, not compiled, not run on hardware
//
#include <hip/hip_runtime.h>

typedef __attribute__((ext_vector_type(16))) __bf16 v16bf;
typedef __attribute__((ext_vector_type(8)))  float  v8f;

#define T_STEPS 4096
#define BB 32
#define DD 256
#define HH 256
#define G3 768              // 3*H
#define HROW 264            // padded h row stride (bf16 elems): 528B = 33*16B
#define HBUF (BB*HROW)      // one h buffer, bf16 elems (8448)
#define XGT (BB*G3)         // 24576 floats per timestep
#define XOFF (2*HBUF*2)     // byte offset of xg buffers in LDS (33792)

union FragU { uint4 u4[2]; v16bf bv; };

static __device__ __forceinline__ v8f wmma_bf16(v16bf a, v16bf b, v8f c) {
    return __builtin_amdgcn_wmma_f32_16x16x32_bf16(false, a, false, b, (short)0, c, false, false);
}

static __device__ __forceinline__ float sigmoidf(float x) {
    return 1.0f / (1.0f + __expf(-x));
}

// ---------------- fp32 -> bf16 elementwise convert ----------------
__global__ void cvt_bf16_kernel(const float* __restrict__ src, __bf16* __restrict__ dst, int n) {
    int i = blockIdx.x * blockDim.x + threadIdx.x;
    if (i < n) dst[i] = (__bf16)src[i];
}

// ---------------- pack W (768 x 256, row-major, fp32) into WMMA B-fragments ----------------
// B[k][n] = W[n][k].  dword index d = ((nt*8 + kt)*32 + lane)*8 + v
// K = 2v + (v>=4 ? 8 : 0) + (lane>=16 ? 8 : 0), +1      N = nt*16 + lane%16
__global__ void pack_w_kernel(const float* __restrict__ W, __bf16* __restrict__ out) {
    int d = blockIdx.x * blockDim.x + threadIdx.x;   // 98304 total
    if (d >= 48 * 8 * 32 * 8) return;
    int v    = d & 7;
    int lane = (d >> 3) & 31;
    int kt   = (d >> 8) & 7;
    int nt   = d >> 11;
    int n  = nt * 16 + (lane & 15);
    int k0 = kt * 32 + 2 * v + ((v < 4) ? 0 : 8) + ((lane < 16) ? 0 : 8);
    out[2 * d + 0] = (__bf16)W[n * DD + k0];
    out[2 * d + 1] = (__bf16)W[n * DD + k0 + 1];
}

// ---------------- Phase 1: xg = x @ W_ih^T + b   (M=131072, N=768, K=256) ----------------
// grid.x = m-tile (8192), 512 threads = 16 waves, wave w handles n-tiles {w, w+16, w+32}
__global__ __launch_bounds__(512) void gemm_xg_kernel(const __bf16* __restrict__ X,
                                                      const __bf16* __restrict__ Wf,
                                                      const float* __restrict__ bias,
                                                      float* __restrict__ XG) {
    const int mt   = blockIdx.x;
    const int wv   = threadIdx.x >> 5;
    const int lane = threadIdx.x & 31;
    const int l16  = lane & 15;
    const int kof  = (lane < 16) ? 0 : 8;
    const int bofs = (lane < 16) ? 0 : 8;

    // A fragments for this m-tile (16 rows x K=256), documented 16-bit A layout
    v16bf a[8];
    const __bf16* xrow = X + (size_t)(mt * 16 + l16) * DD;
#pragma unroll
    for (int kt = 0; kt < 8; ++kt) {
        FragU f;
        f.u4[0] = *(const uint4*)(xrow + kt * 32 + kof);
        f.u4[1] = *(const uint4*)(xrow + kt * 32 + 16 + kof);
        a[kt] = f.bv;
    }

#pragma unroll
    for (int g = 0; g < 3; ++g) {
        const int nt = g * 16 + wv;
        const float bv = bias[nt * 16 + l16];
        v8f acc = {bv, bv, bv, bv, bv, bv, bv, bv};
#pragma unroll
        for (int kt = 0; kt < 8; ++kt) {
            v16bf bf = *(const v16bf*)(Wf + ((size_t)(nt * 8 + kt) * 32 + lane) * 16);
            acc = wmma_bf16(a[kt], bf, acc);
        }
#pragma unroll
        for (int v = 0; v < 8; ++v) {
            int m = mt * 16 + v + bofs;
            XG[(size_t)m * G3 + nt * 16 + l16] = acc[v];
        }
    }
}

// ---------------- Phase 2: sequential GRU scan, 1 workgroup, 16 waves ----------------
// W_hh B-fragments register-resident (393KB > 320KB LDS).  h double-buffered in LDS (bf16).
// xg_t double-buffered in LDS (fp32, 2x96KB), prefetched with global_load_async_to_lds_b128.
// All LDS addressing is done with integer offsets off a single addrspace(3) base so the
// compiler emits real ds_* ops (pointer arrays previously degraded these to FLAT).
__global__ __launch_bounds__(512) void gru_scan_kernel(const float* __restrict__ XG,
                                                       const float* __restrict__ h0,
                                                       const __bf16* __restrict__ Wf,
                                                       const float* __restrict__ b_n,
                                                       float* __restrict__ out) {
    extern __shared__ char smem[];
    __bf16* hbase = (__bf16*)smem;            // [2][BB][HROW] bf16
    float*  xbase = (float*)(smem + XOFF);    // [2][BB][G3]   fp32

    const int tid  = threadIdx.x;
    const int wv   = tid >> 5;
    const int lane = tid & 31;
    const int l16  = lane & 15;
    const int bofs = (lane < 16) ? 0 : 8;
    const int jj   = wv * 16 + l16;        // hidden index owned by this wave/lane column

    // Load this wave's W_hh B-fragments into registers (stay live across all 4096 steps)
    v16bf bw[3][8];
#pragma unroll
    for (int g = 0; g < 3; ++g) {
        const int nt = g * 16 + wv;
#pragma unroll
        for (int kt = 0; kt < 8; ++kt)
            bw[g][kt] = *(const v16bf*)(Wf + ((size_t)(nt * 8 + kt) * 32 + lane) * 16);
    }
    const float bn = b_n[jj];

    // init h (bf16) into buffer 0
    for (int i = tid; i < BB * HH; i += 512) {
        int b = i >> 8, j = i & 255;
        hbase[b * HROW + j] = (__bf16)h0[i];
    }
    // synchronous preload of xg_0
    for (int i = tid; i < XGT; i += 512) xbase[i] = XG[i];
    __syncthreads();

    // LDS byte address of the xg buffers (for the async-to-LDS engine)
    const unsigned xlds0 = (unsigned)(size_t)(void*)xbase;

    for (int t = 0; t < T_STEPS; ++t) {
        const int cur = t & 1, nxt = cur ^ 1;
        const int hC = cur * HBUF, hN = nxt * HBUF;   // h buffer offsets (bf16 elems)
        const int xC = cur * XGT;                     // xg buffer offset (f32 elems)

        // --- async prefetch xg_{t+1} into the other LDS buffer (ASYNCcnt path) ---
        if (t + 1 < T_STEPS) {
            const char* gsrc = (const char*)(XG + (size_t)(t + 1) * XGT);
            const unsigned xlds = xlds0 + (unsigned)nxt * (XGT * 4);
#pragma unroll
            for (int c = 0; c < 12; ++c) {
                unsigned off = (unsigned)(tid * 16 + c * 8192);   // 512 thr * 16B chunks
                asm volatile("global_load_async_to_lds_b128 %0, %1, %2"
                             :
                             : "v"(xlds + off), "v"(off), "s"(gsrc)
                             : "memory");
            }
        }

        // --- hg = h @ W_hh^T via WMMA; r/z accumulators seeded with xg (free bias/xg add) ---
        v8f acc[2][3];
#pragma unroll
        for (int mt = 0; mt < 2; ++mt)
#pragma unroll
            for (int g = 0; g < 3; ++g)
#pragma unroll
                for (int v = 0; v < 8; ++v) {
                    int bi = mt * 16 + v + bofs;
                    acc[mt][g][v] = (g < 2) ? xbase[xC + bi * G3 + g * HH + jj] : 0.0f;
                }

#pragma unroll
        for (int kt = 0; kt < 8; ++kt) {
#pragma unroll
            for (int mt = 0; mt < 2; ++mt) {
                const __bf16* hrow = hbase + hC + (mt * 16 + l16) * HROW + kt * 32 + bofs;
                FragU f;
                f.u4[0] = *(const uint4*)(hrow);
                f.u4[1] = *(const uint4*)(hrow + 16);
                v16bf a = f.bv;
#pragma unroll
                for (int g = 0; g < 3; ++g)
                    acc[mt][g] = wmma_bf16(a, bw[g][kt], acc[mt][g]);
            }
        }

        // --- gates + state update (each wave owns its 16-wide hidden slice, all 32 batches) ---
#pragma unroll
        for (int mt = 0; mt < 2; ++mt)
#pragma unroll
            for (int v = 0; v < 8; ++v) {
                int bi = mt * 16 + v + bofs;
                float r  = sigmoidf(acc[mt][0][v]);
                float z  = sigmoidf(acc[mt][1][v]);
                float xn = xbase[xC + bi * G3 + 2 * HH + jj];
                float nn = tanhf(xn + r * (acc[mt][2][v] + bn));
                float hp = (float)hbase[hC + bi * HROW + jj];
                float hnew = nn + z * (hp - nn);
                hbase[hN + bi * HROW + jj] = (__bf16)hnew;
                out[(size_t)t * (BB * HH) + bi * HH + jj] = hnew;
            }

        asm volatile("s_wait_asynccnt 0x0" ::: "memory");
        __syncthreads();
    }
}

extern "C" void kernel_launch(void* const* d_in, const int* in_sizes, int n_in,
                              void* d_out, int out_size, void* d_ws, size_t ws_size,
                              hipStream_t stream) {
    const float* x    = (const float*)d_in[0];   // (T,B,D)
    const float* h0   = (const float*)d_in[1];   // (B,H)
    const float* Wih  = (const float*)d_in[2];   // (3H,D)
    const float* Whh  = (const float*)d_in[3];   // (3H,H)
    const float* bias = (const float*)d_in[4];   // (3H,)
    const float* bn   = (const float*)d_in[5];   // (H,)
    float* out = (float*)d_out;

    char* ws = (char*)d_ws;
    __bf16* xbf  = (__bf16*)ws;                             // 67,108,864 B
    __bf16* wihf = (__bf16*)(ws + 67108864);                //    393,216 B
    __bf16* whhf = (__bf16*)(ws + 67108864 + 393216);       //    393,216 B
    float*  xg   = (float*)(ws + 67108864 + 2 * 393216);    // 402,653,184 B

    const int nX = T_STEPS * BB * DD;                       // 33,554,432
    cvt_bf16_kernel<<<nX / 256, 256, 0, stream>>>(x, xbf, nX);
    pack_w_kernel<<<384, 256, 0, stream>>>(Wih, wihf);
    pack_w_kernel<<<384, 256, 0, stream>>>(Whh, whhf);

    gemm_xg_kernel<<<(T_STEPS * BB) / 16, 512, 0, stream>>>(xbf, wihf, bias, xg);

    size_t lds_bytes = XOFF + 2 * XGT * 4;                  // 33,792 + 196,608 = 230,400 B
    gru_scan_kernel<<<1, 512, lds_bytes, stream>>>(xg, h0, whhf, bn, out);
}
